// CausalSelfAttention_66383014527138
// MI455X (gfx1250) — matrix operand order repaired and checked by reading, not compiled, not run on hardware
//
#include <hip/hip_runtime.h>
#include <stdint.h>

// ---------------------------------------------------------------------------
// Causal self-attention forward, bf16 WMMA pipeline for gfx1250 (wave32).
//   B=4, T=2048, C=1024, H=16, D=64
// GEMMs use async global->LDS tile staging (ASYNCcnt) with double buffering.
// ---------------------------------------------------------------------------

typedef __attribute__((ext_vector_type(16))) __bf16 v16bf;
typedef __attribute__((ext_vector_type(8)))  float  v8f;

union Frag {
  v16bf v;
  uint4 u[2];
};

static constexpr int B_ = 4, T_ = 2048, C_ = 1024, H_ = 16, D_ = 64;
static constexpr int M_ = B_ * T_;     // 8192 rows (tokens)
static constexpr int NQKV_ = 3 * C_;   // 3072
static constexpr int LDSP = 40;        // padded LDS row stride (bf16 elems, 80B)

__device__ __forceinline__ unsigned short f2bf(float f) {
  unsigned int u = __builtin_bit_cast(unsigned int, f);
  u += 0x7fffu + ((u >> 16) & 1u);   // round-to-nearest-even
  return (unsigned short)(u >> 16);
}

__device__ __forceinline__ v8f zero8() {
  v8f z;
  #pragma unroll
  for (int i = 0; i < 8; ++i) z[i] = 0.0f;
  return z;
}

__device__ __forceinline__ v8f wmma_bf16(const Frag& a, const Frag& b, v8f c) {
  return __builtin_amdgcn_wmma_f32_16x16x32_bf16(
      false, a.v, false, b.v, (short)0, c, false, false);
}

// Flat LDS aperture keeps the 0-based LDS byte offset in the low 32 bits.
__device__ __forceinline__ unsigned lds_off(const void* p) {
  return (unsigned)(uintptr_t)p;
}

// Async copy of 32 bytes global -> LDS (two b128), no VGPR staging.
// GVS mode: SGPR 64-bit base + per-lane 32-bit byte offset; instruction
// offset applies to both LDS and global addresses.
__device__ __forceinline__ void async_copy32(unsigned lds_byte,
                                             unsigned gbyte_off,
                                             const void* sbase) {
  asm volatile(
      "global_load_async_to_lds_b128 %0, %1, %2\n\t"
      "global_load_async_to_lds_b128 %0, %1, %2 offset:16"
      :
      : "v"(lds_byte), "v"(gbyte_off), "s"(sbase)
      : "memory");
}

__device__ __forceinline__ void wait_async0() {
  asm volatile("s_wait_asynccnt 0x0" ::: "memory");
}

// --------------------------- conversion kernels ----------------------------

__global__ void cvt_bf16_kernel(const float* __restrict__ src,
                                unsigned short* __restrict__ dst, int n) {
  int i = blockIdx.x * blockDim.x + threadIdx.x;
  if (i < n) dst[i] = f2bf(src[i]);
}

// dst[N][K] = bf16(src[K][N])   (weights -> N-major so B-fragments are linear)
__global__ void transpose_bf16_kernel(const float* __restrict__ src,
                                      unsigned short* __restrict__ dst,
                                      int K, int N) {
  long i = (long)blockIdx.x * blockDim.x + threadIdx.x;
  if (i >= (long)K * N) return;
  int n = (int)(i / K);
  int k = (int)(i % K);
  dst[i] = f2bf(src[(long)k * N + n]);
}

// ------------------------------ tiled GEMM ---------------------------------
// C[M][N] = A[M][K] * Bt[N][K]^T, 128x128 block tile, BK=32, 8 waves,
// each wave: 32x64 (2x4 WMMA 16x16 tiles). Double-buffered async LDS staging.
// MODE 0: scatter epilogue into Q[B,H,T,D], K[B,H,T,D], Vt[B,H,D,T] (bf16)
// MODE 1: plain fp32 row-major output (N=1024)

template <int MODE>
__global__ __launch_bounds__(256)
void gemm_bf16_kernel(const unsigned short* __restrict__ A,
                      const unsigned short* __restrict__ Bt,
                      int Kdim,
                      float* __restrict__ outF,
                      unsigned short* __restrict__ qb,
                      unsigned short* __restrict__ kb,
                      unsigned short* __restrict__ vt) {
  __shared__ unsigned short lsA[2][128 * LDSP];
  __shared__ unsigned short lsB[2][128 * LDSP];

  const int tid  = threadIdx.x;
  const int lane = tid & 31;
  const int wave = tid >> 5;
  const int half = lane >> 4;
  const int l15  = lane & 15;
  const int wm   = (wave >> 1) * 32;  // wave row base within block tile
  const int wn   = (wave & 1) * 64;   // wave col base within block tile
  const int m0   = blockIdx.y * 128;
  const int n0   = blockIdx.x * 128;

  v8f acc[2][4];
  #pragma unroll
  for (int mt = 0; mt < 2; ++mt)
    #pragma unroll
    for (int nt = 0; nt < 4; ++nt) acc[mt][nt] = zero8();

  const int ldrow   = tid >> 1;        // 0..127
  const int ldchunk = (tid & 1) * 16;  // 0 or 16 (bf16 elems)

  // Per-thread staging addresses (bytes).
  const unsigned gA0 = (unsigned)(((m0 + ldrow) * Kdim + ldchunk) * 2);
  const unsigned gB0 = (unsigned)(((n0 + ldrow) * Kdim + ldchunk) * 2);
  unsigned lA[2], lB[2];
  #pragma unroll
  for (int bfi = 0; bfi < 2; ++bfi) {
    lA[bfi] = lds_off(&lsA[bfi][ldrow * LDSP + ldchunk]);
    lB[bfi] = lds_off(&lsB[bfi][ldrow * LDSP + ldchunk]);
  }

  // Prologue: stage k-slab 0 into buffer 0.
  async_copy32(lA[0], gA0, A);
  async_copy32(lB[0], gB0, Bt);

  for (int k0 = 0; k0 < Kdim; k0 += 32) {
    const int cur = (k0 >> 5) & 1;
    wait_async0();       // this slab's async fills are done (per-wave)
    __syncthreads();     // all waves' fills visible; prior reads drained

    if (k0 + 32 < Kdim) {  // overlap next slab's fill with this slab's math
      const unsigned koff = (unsigned)((k0 + 32) * 2);
      async_copy32(lA[cur ^ 1], gA0 + koff, A);
      async_copy32(lB[cur ^ 1], gB0 + koff, Bt);
    }

    Frag af[2], bfr[4];
    #pragma unroll
    for (int mt = 0; mt < 2; ++mt) {
      // A 16x32 bf16 layout: lanes 0-15 K={0..7,16..23}, lanes 16-31 K={8..15,24..31}
      const unsigned short* p = &lsA[cur][(wm + mt * 16 + l15) * LDSP];
      af[mt].u[0] = *(const uint4*)(p + half * 8);
      af[mt].u[1] = *(const uint4*)(p + 16 + half * 8);
    }
    #pragma unroll
    for (int nt = 0; nt < 4; ++nt) {
      // B 32x16 bf16 layout: lanes 0-15 K={0..7,16..23}, lanes 16-31 K={8..15,24..31}, N=lane&15
      const unsigned short* p = &lsB[cur][(wn + nt * 16 + l15) * LDSP];
      bfr[nt].u[0] = *(const uint4*)(p + half * 8);
      bfr[nt].u[1] = *(const uint4*)(p + 16 + half * 8);
    }
    #pragma unroll
    for (int mt = 0; mt < 2; ++mt)
      #pragma unroll
      for (int nt = 0; nt < 4; ++nt)
        acc[mt][nt] = wmma_bf16(af[mt], bfr[nt], acc[mt][nt]);
  }

  // epilogue.  C layout: VGPR r -> row = r + 8*half, col = lane&15
  #pragma unroll
  for (int mt = 0; mt < 2; ++mt)
    #pragma unroll
    for (int nt = 0; nt < 4; ++nt)
      #pragma unroll
      for (int r = 0; r < 8; ++r) {
        int mg = m0 + wm + mt * 16 + r + half * 8;
        int ng = n0 + wn + nt * 16 + l15;
        float val = acc[mt][nt][r];
        if (MODE == 1) {
          outF[(long)mg * C_ + ng] = val;
        } else {
          int b = mg >> 11, t = mg & (T_ - 1);
          unsigned short bv = f2bf(val);
          if (ng < C_) {
            int h = ng >> 6, d = ng & 63;
            qb[(((long)b * H_ + h) * T_ + t) * D_ + d] = bv;
          } else if (ng < 2 * C_) {
            int nn = ng - C_;
            int h = nn >> 6, d = nn & 63;
            kb[(((long)b * H_ + h) * T_ + t) * D_ + d] = bv;
          } else {
            int nn = ng - 2 * C_;
            int h = nn >> 6, d = nn & 63;
            vt[(((long)b * H_ + h) * D_ + d) * T_ + t] = bv;  // V stored transposed
          }
        }
      }
}

// --------------------------- flash attention -------------------------------
// Each wave owns 16 query rows; online softmax over causal key chunks of 32.
// S = Q K^T via 4 WMMA / chunk, P V via 4 WMMA / chunk. P is re-shaped
// C-layout -> A-layout through a per-wave LDS patch. K/V stream from global
// (per-head working set is 256KB -> L2 resident; 192MB L2).

__global__ __launch_bounds__(256)
void attn_kernel(const unsigned short* __restrict__ qb,
                 const unsigned short* __restrict__ kb,
                 const unsigned short* __restrict__ vt,
                 unsigned short* __restrict__ yb) {
  __shared__ unsigned short lsP[8 * 16 * LDSP];

  const int tid  = threadIdx.x;
  const int lane = tid & 31;
  const int wave = tid >> 5;
  const int half = lane >> 4;
  const int l15  = lane & 15;
  const int bh   = blockIdx.y;           // b*H + h
  const int b    = bh >> 4, h = bh & 15;
  const int qw   = blockIdx.x * 128 + wave * 16;  // first query row of this wave

  const unsigned short* qhead = qb + ((long)bh * T_ + qw) * D_;
  const unsigned short* khead = kb + (long)bh * T_ * D_;
  const unsigned short* vhead = vt + (long)bh * D_ * T_;

  // Q A-fragments (16 queries x 64 d  ->  two 16x32 fragments), loop-invariant
  Frag aq[2];
  #pragma unroll
  for (int s = 0; s < 2; ++s) {
    const unsigned short* p = qhead + l15 * D_ + s * 32;
    aq[s].u[0] = *(const uint4*)(p + half * 8);
    aq[s].u[1] = *(const uint4*)(p + 16 + half * 8);
  }

  v8f yacc[4];
  #pragma unroll
  for (int j = 0; j < 4; ++j) yacc[j] = zero8();
  float rm[8], rs[8];
  #pragma unroll
  for (int r = 0; r < 8; ++r) { rm[r] = -3.0e38f; rs[r] = 0.0f; }

  unsigned short* pls = lsP + wave * 16 * LDSP;
  const float scale = 0.125f;  // 1/sqrt(64)

  for (int kc = 0; kc < qw + 16; kc += 32) {
    // ---- S tile: 16 queries x 32 keys ----
    v8f s0 = zero8(), s1 = zero8();
    #pragma unroll
    for (int s = 0; s < 2; ++s) {
      Frag bk0, bk1;
      const unsigned short* p0 = khead + (long)(kc + l15) * D_ + s * 32 + half * 8;
      bk0.u[0] = *(const uint4*)(p0);
      bk0.u[1] = *(const uint4*)(p0 + 16);
      const unsigned short* p1 = khead + (long)(kc + 16 + l15) * D_ + s * 32 + half * 8;
      bk1.u[0] = *(const uint4*)(p1);
      bk1.u[1] = *(const uint4*)(p1 + 16);
      s0 = wmma_bf16(aq[s], bk0, s0);
      s1 = wmma_bf16(aq[s], bk1, s1);
    }

    // ---- causal mask + online softmax (row stats replicated per half) ----
    #pragma unroll
    for (int r = 0; r < 8; ++r) {
      int qrow = qw + r + half * 8;
      float v0 = (kc + l15 <= qrow)      ? s0[r] * scale : -1.0e30f;
      float v1 = (kc + 16 + l15 <= qrow) ? s1[r] * scale : -1.0e30f;
      float mx = fmaxf(v0, v1);
      mx = fmaxf(mx, __shfl_xor(mx, 1, 32));
      mx = fmaxf(mx, __shfl_xor(mx, 2, 32));
      mx = fmaxf(mx, __shfl_xor(mx, 4, 32));
      mx = fmaxf(mx, __shfl_xor(mx, 8, 32));
      float mnew  = fmaxf(rm[r], mx);
      float alpha = __expf(rm[r] - mnew);
      float p0f = __expf(v0 - mnew);
      float p1f = __expf(v1 - mnew);
      float ps = p0f + p1f;
      ps += __shfl_xor(ps, 1, 32);
      ps += __shfl_xor(ps, 2, 32);
      ps += __shfl_xor(ps, 4, 32);
      ps += __shfl_xor(ps, 8, 32);
      rs[r] = rs[r] * alpha + ps;
      rm[r] = mnew;
      #pragma unroll
      for (int j = 0; j < 4; ++j) yacc[j][r] *= alpha;
      int m = r + half * 8;
      pls[m * LDSP + l15]      = f2bf(p0f);
      pls[m * LDSP + 16 + l15] = f2bf(p1f);
    }

    // ---- P (16x32) as A-fragment; accumulate P*V over 4 d-tiles ----
    Frag ap;
    const unsigned short* pp = pls + l15 * LDSP;
    ap.u[0] = *(const uint4*)(pp + half * 8);
    ap.u[1] = *(const uint4*)(pp + 16 + half * 8);
    #pragma unroll
    for (int j = 0; j < 4; ++j) {
      Frag bv;
      const unsigned short* pv = vhead + (long)(j * 16 + l15) * T_ + kc + half * 8;
      bv.u[0] = *(const uint4*)(pv);
      bv.u[1] = *(const uint4*)(pv + 16);
      yacc[j] = wmma_bf16(ap, bv, yacc[j]);
    }
  }

  // ---- normalize and write y (bf16, token-major [B*T][C]) ----
  #pragma unroll
  for (int r = 0; r < 8; ++r) {
    float inv = 1.0f / rs[r];
    int t = qw + r + half * 8;
    #pragma unroll
    for (int j = 0; j < 4; ++j)
      yb[((long)b * T_ + t) * C_ + h * D_ + j * 16 + l15] = f2bf(yacc[j][r] * inv);
  }
}

// ------------------------------- launcher ----------------------------------

extern "C" void kernel_launch(void* const* d_in, const int* in_sizes, int n_in,
                              void* d_out, int out_size, void* d_ws, size_t ws_size,
                              hipStream_t stream) {
  const float* x      = (const float*)d_in[0];  // [B,T,C]
  const float* w_qkv  = (const float*)d_in[1];  // [C,3C]
  const float* w_proj = (const float*)d_in[2];  // [C,C]
  float* out = (float*)d_out;                   // [B,T,C] fp32

  char* ws = (char*)d_ws;
  size_t off = 0;
  auto carve = [&](size_t bytes) {
    char* p = ws + off;
    off += (bytes + 255) & ~(size_t)255;
    return p;
  };
  unsigned short* xb     = (unsigned short*)carve((size_t)M_ * C_ * 2);
  unsigned short* wqkvT  = (unsigned short*)carve((size_t)NQKV_ * C_ * 2);
  unsigned short* wprojT = (unsigned short*)carve((size_t)C_ * C_ * 2);
  unsigned short* qb     = (unsigned short*)carve((size_t)M_ * C_ * 2);
  unsigned short* kb     = (unsigned short*)carve((size_t)M_ * C_ * 2);
  unsigned short* vt     = (unsigned short*)carve((size_t)M_ * C_ * 2);
  unsigned short* yb     = (unsigned short*)carve((size_t)M_ * C_ * 2);

  // 1) x -> bf16
  cvt_bf16_kernel<<<(M_ * C_ + 255) / 256, 256, 0, stream>>>(x, xb, M_ * C_);
  // 2) weights -> bf16, transposed to N-major
  transpose_bf16_kernel<<<((long)NQKV_ * C_ + 255) / 256, 256, 0, stream>>>(
      w_qkv, wqkvT, C_, NQKV_);
  transpose_bf16_kernel<<<((long)C_ * C_ + 255) / 256, 256, 0, stream>>>(
      w_proj, wprojT, C_, C_);
  // 3) qkv = x @ w_qkv, scattered into Q/K/Vt head layouts
  gemm_bf16_kernel<0><<<dim3(NQKV_ / 128, M_ / 128), 256, 0, stream>>>(
      xb, wqkvT, C_, nullptr, qb, kb, vt);
  // 4) flash attention
  attn_kernel<<<dim3(T_ / 128, B_ * H_), 256, 0, stream>>>(qb, kb, vt, yb);
  // 5) out = y @ w_proj  (fp32 output)
  gemm_bf16_kernel<1><<<dim3(C_ / 128, M_ / 128), 256, 0, stream>>>(
      yb, wprojT, C_, out, nullptr, nullptr, nullptr);
}
